// EquivariantMessagePassing_82867099009204
// MI455X (gfx1250) — matrix operand order repaired and checked by reading, not compiled, not run on hardware
//
#include <hip/hip_runtime.h>
#include <hip/hip_bf16.h>

// ---------------------------------------------------------------------------
// EGNN message passing on gfx1250 (MI455X).
// Both MLP layers run on the WMMA pipe: v_wmma_f32_16x16x32_bf16 (bf16 in,
// fp32 accumulate). dist_sq / biases / SiLU / scatter-adds stay fp32.
//
// B-stationary schedule: one wave owns T=4 WMMA row-tiles (64 edges) and
// reuses every weight (B) fragment across 4 matrix ops, cutting weight-side
// cache traffic 4x (the only stream that exceeded the compute roofline).
// The x-row gather goes memory->LDS via the gfx1250 async-to-LDS path
// (global_load_async_to_lds_b128, tracked by ASYNCcnt).
// SiLU uses v_exp_f32 + v_rcp_f32 (1-ulp) instead of an IEEE divide chain.
// ---------------------------------------------------------------------------

typedef __bf16  bf16s;
typedef bf16s   v16bf __attribute__((ext_vector_type(16)));
typedef float   v8f   __attribute__((ext_vector_type(8)));

union Frag16 { uint4 q[2]; v16bf v; unsigned short s16[16]; };

__device__ __forceinline__ unsigned short f2bf(float f) {
    union { bf16s b; unsigned short s; } u;
    u.b = (bf16s)f;
    return u.s;
}

__device__ __forceinline__ float fast_silu(float h) {
    const float den = 1.0f + __expf(-h);
#if defined(__has_builtin)
#if __has_builtin(__builtin_amdgcn_rcpf)
    return h * __builtin_amdgcn_rcpf(den);      // v_rcp_f32, ~1 ulp
#else
    return h / den;
#endif
#else
    return h / den;
#endif
}

#define F_DIM 64
#define TTILE 4          // 16-edge WMMA tiles per wave -> 64 edges per wave

#if defined(__has_builtin)
#if __has_builtin(__builtin_amdgcn_global_load_async_to_lds_b128)
#define HAVE_ASYNC_LDS 1
#endif
#endif

__device__ __forceinline__ void async_copy16(const float* g, float* l) {
#ifdef HAVE_ASYNC_LDS
    // builtin signature (from hipcc diagnostic): param0 is v4i32 AS1*, so
    // pass address-space-qualified vector pointers.
    typedef int vi4 __attribute__((vector_size(16)));
    typedef vi4 __attribute__((address_space(1)))* gptr_t;
    typedef vi4 __attribute__((address_space(3)))* lptr_t;
    __builtin_amdgcn_global_load_async_to_lds_b128((gptr_t)g, (lptr_t)l, 0, 0);
#else
    *(float4*)l = *(const float4*)g;
#endif
}

__device__ __forceinline__ void wait_async_done() {
#ifdef HAVE_ASYNC_LDS
#if __has_builtin(__builtin_amdgcn_s_wait_asynccnt)
    __builtin_amdgcn_s_wait_asynccnt(0);
    asm volatile("" ::: "memory");
#else
    asm volatile("s_wait_asynccnt 0x0" ::: "memory");
#endif
#endif
}

// ---------------------------------------------------------------------------
// Weight packing kernels
// ---------------------------------------------------------------------------
__global__ void pack_w1(const float* __restrict__ W1x, const float* __restrict__ W1p,
                        const float* __restrict__ b1x, const float* __restrict__ b1p,
                        unsigned short* __restrict__ W1bf,
                        float* __restrict__ w1last, float* __restrict__ b1cat) {
    int idx = blockIdx.x * blockDim.x + threadIdx.x;       // 129*256 threads
    if (idx < 129 * 256) {
        int k = idx >> 8, n = idx & 255;
        float v = (n < 128) ? W1x[k * 128 + n] : W1p[k * 128 + (n - 128)];
        if (k < 128) W1bf[k * 256 + n] = f2bf(v);
        else         w1last[n] = v;                        // dist_sq row stays fp32
    }
    if (idx < 256) b1cat[idx] = (idx < 128) ? b1x[idx] : b1p[idx - 128];
}

__global__ void pack_w2(const float* __restrict__ W2x, const float* __restrict__ W2p,
                        const float* __restrict__ b2x, const float* __restrict__ b2p,
                        unsigned short* __restrict__ W2bf, float* __restrict__ b2cat) {
    int idx = blockIdx.x * blockDim.x + threadIdx.x;       // 256*80 threads
    if (idx < 256 * 80) {
        int k = idx / 80, n = idx % 80;
        float v = 0.0f;
        if (k < 128) { if (n < 64)  v = W2x[k * 64 + n]; }          // x-branch block
        else         { if (n == 64) v = W2p[k - 128]; }             // p-branch block
        W2bf[idx] = f2bf(v);
    }
    if (idx < 80)
        b2cat[idx] = (idx < 64) ? b2x[idx] : ((idx == 64) ? b2p[0] : 0.0f);
}

__global__ void zero_f32(float* __restrict__ p, int n) {
    int i = blockIdx.x * blockDim.x + threadIdx.x;
    if (i < n) p[i] = 0.0f;
}

// ---------------------------------------------------------------------------
// Main kernel: one wave (single-wave workgroup) handles 4 x 16-edge tiles
// ---------------------------------------------------------------------------
__global__ __launch_bounds__(32)
void egnn_kernel(const float* __restrict__ x, const float* __restrict__ pos,
                 const int* __restrict__ eidx, int E, int N,
                 const unsigned short* __restrict__ W1bf,
                 const float* __restrict__ w1last, const float* __restrict__ b1cat,
                 const unsigned short* __restrict__ W2bf,
                 const float* __restrict__ b2cat,
                 float* __restrict__ outx, float* __restrict__ outp) {
    __shared__ __align__(16) float          sCombF[16][132];       // fp32 staging, 1 tile
    __shared__ __align__(16) unsigned short sHid[TTILE][16][264];  // bf16 silu(hidden)
    __shared__ float4 sRel[TTILE * 16];                            // rel.xyz, dist_sq
    __shared__ int    sRowA[TTILE * 16], sColA[TTILE * 16], sValA[TTILE * 16];

    const int lane = threadIdx.x & 31;
    const int half = lane >> 4;
    const int l16  = lane & 15;
    const int ebase = blockIdx.x * (TTILE * 16);

    // ---- stage 1: indices, rel_pos, dist_sq for 64 edges ------------------
    #pragma unroll
    for (int rep = 0; rep < 2; ++rep) {
        const int sl = lane + 32 * rep;                 // edge slot 0..63
        const int e  = ebase + sl;
        const int valid = (e < E) ? 1 : 0;
        int r = valid ? eidx[e]     : 0;
        int c = valid ? eidx[E + e] : 0;
        r = (r < 0 || r >= N) ? 0 : r;
        c = (c < 0 || c >= N) ? 0 : c;
        const float rx = pos[r * 3 + 0] - pos[c * 3 + 0];
        const float ry = pos[r * 3 + 1] - pos[c * 3 + 1];
        const float rz = pos[r * 3 + 2] - pos[c * 3 + 2];
        sRel[sl]  = make_float4(rx, ry, rz, rx * rx + ry * ry + rz * rz);
        sRowA[sl] = r;
        sColA[sl] = c;
        sValA[sl] = valid;
    }
    __syncthreads();

    // ---- stage 2: per tile, gather x rows (async->LDS), build A1 frags ----
    // A fragment (16x32 bf16): lane (l16,half) holds row M=l16,
    //   elems 0..7  -> K = kk*32 + 8*half .. +7
    //   elems 8..15 -> K = kk*32 + 16 + 8*half .. +7
    Frag16 a1[TTILE][4];                                // 128 VGPRs, B-reuse enabler
    #pragma unroll
    for (int t = 0; t < TTILE; ++t) {
        // lanes 0..15 copy x[row] (16B each), lanes 16..31 copy x[col]
        #pragma unroll 4
        for (int i = 0; i < 16; ++i) {
            const int sl  = t * 16 + i;
            const int src = half ? sColA[sl] : sRowA[sl];
            async_copy16(x + (size_t)src * F_DIM + 4 * l16,
                         &sCombF[i][half * 64 + 4 * l16]);
        }
        wait_async_done();
        __syncthreads();
        #pragma unroll
        for (int kk = 0; kk < 4; ++kk) {
            const float* p = &sCombF[l16][kk * 32 + 8 * half];
            const float4 f0 = *(const float4*)(p);
            const float4 f1 = *(const float4*)(p + 4);
            const float4 f2 = *(const float4*)(p + 16);
            const float4 f3 = *(const float4*)(p + 20);
            Frag16 a;
            a.s16[0]  = f2bf(f0.x); a.s16[1]  = f2bf(f0.y);
            a.s16[2]  = f2bf(f0.z); a.s16[3]  = f2bf(f0.w);
            a.s16[4]  = f2bf(f1.x); a.s16[5]  = f2bf(f1.y);
            a.s16[6]  = f2bf(f1.z); a.s16[7]  = f2bf(f1.w);
            a.s16[8]  = f2bf(f2.x); a.s16[9]  = f2bf(f2.y);
            a.s16[10] = f2bf(f2.z); a.s16[11] = f2bf(f2.w);
            a.s16[12] = f2bf(f3.x); a.s16[13] = f2bf(f3.y);
            a.s16[14] = f2bf(f3.z); a.s16[15] = f2bf(f3.w);
            a1[t][kk] = a;
        }
        __syncthreads();                                // before buffer reuse
    }

    float dq[TTILE][8];                                 // dist_sq per output row
    #pragma unroll
    for (int t = 0; t < TTILE; ++t)
        #pragma unroll
        for (int i = 0; i < 8; ++i) dq[t][i] = sRel[t * 16 + i + 8 * half].w;

    // ---- stage 3: hidden = silu(combined @ [W1x|W1p] + b1 + dsq*w1_last) --
    // B-stationary: each B fragment feeds 4 WMMAs (one per edge tile).
    for (int nt = 0; nt < 16; ++nt) {                   // 256 hidden cols / 16
        const int col  = nt * 16 + l16;
        const float bias = b1cat[col];
        const float wl   = w1last[col];
        v8f acc[TTILE];
        #pragma unroll
        for (int t = 0; t < TTILE; ++t)
            #pragma unroll
            for (int i = 0; i < 8; ++i) acc[t][i] = fmaf(dq[t][i], wl, bias);
        #pragma unroll
        for (int kk = 0; kk < 4; ++kk) {                // K = 128
            // B fragment (32x16 bf16): lane holds col N = nt*16 + l16,
            //   elems 0..7 -> K = kk*32 + 8*half .. +7, elems 8..15 -> K = kk*32 + 16 + 8*half .. +7
            Frag16 b;
            const unsigned short* bp =
                W1bf + (kk * 32 + 8 * half) * 256 + nt * 16 + l16;
            #pragma unroll
            for (int s = 0; s < 8; ++s) {
                b.s16[s]     = bp[s * 256];
                b.s16[8 + s] = bp[(16 + s) * 256];
            }
            #pragma unroll
            for (int t = 0; t < TTILE; ++t)
                acc[t] = __builtin_amdgcn_wmma_f32_16x16x32_bf16(
                    false, a1[t][kk].v, false, b.v, (short)0, acc[t], false, false);
        }
        // SiLU in fp32, transpose to LDS as bf16 (D layout: lane=N, VGPR=M)
        #pragma unroll
        for (int t = 0; t < TTILE; ++t)
            #pragma unroll
            for (int i = 0; i < 8; ++i)
                sHid[t][i + 8 * half][nt * 16 + l16] = f2bf(fast_silu(acc[t][i]));
    }
    __syncthreads();

    // ---- stage 4: [msg_x | w_pos] = hidden @ W2cat + b2cat ----------------
    // B-stationary again; A2 fragments re-streamed from LDS (cheap, per-WGP).
    #pragma unroll
    for (int n2 = 0; n2 < 5; ++n2) {                    // 80 cols / 16
        const float bias = b2cat[n2 * 16 + l16];
        v8f acc[TTILE];
        #pragma unroll
        for (int t = 0; t < TTILE; ++t)
            #pragma unroll
            for (int i = 0; i < 8; ++i) acc[t][i] = bias;
        #pragma unroll
        for (int kk = 0; kk < 8; ++kk) {                // K = 256
            Frag16 b;
            const unsigned short* bp =
                W2bf + (kk * 32 + 8 * half) * 80 + n2 * 16 + l16;
            #pragma unroll
            for (int s = 0; s < 8; ++s) {
                b.s16[s]     = bp[s * 80];
                b.s16[8 + s] = bp[(16 + s) * 80];
            }
            #pragma unroll
            for (int t = 0; t < TTILE; ++t) {
                Frag16 a;
                const unsigned short* p = &sHid[t][l16][kk * 32 + 8 * half];
                a.q[0] = *(const uint4*)p;
                a.q[1] = *(const uint4*)(p + 16);
                acc[t] = __builtin_amdgcn_wmma_f32_16x16x32_bf16(
                    false, a.v, false, b.v, (short)0, acc[t], false, false);
            }
        }
        if (n2 < 4) {
            // msg_x scatter: segment_sum over col via fp32 global atomics
            #pragma unroll
            for (int t = 0; t < TTILE; ++t)
                #pragma unroll
                for (int i = 0; i < 8; ++i) {
                    const int m = t * 16 + i + 8 * half;     // edge slot
                    if (sValA[m])
                        atomicAdd(outx + (size_t)sColA[m] * F_DIM + n2 * 16 + l16,
                                  acc[t][i]);
                }
        } else if (l16 == 0) {
            // w_pos lives in packed column 64 (tile 4, N=0): lanes 0 & 16
            #pragma unroll
            for (int t = 0; t < TTILE; ++t)
                #pragma unroll
                for (int i = 0; i < 8; ++i) {
                    const int m = t * 16 + i + 8 * half;
                    if (sValA[m]) {
                        const float4 rel = sRel[m];
                        const float wp = acc[t][i];
                        float* pd = outp + (size_t)sColA[m] * 3;
                        atomicAdd(pd + 0, wp * rel.x);
                        atomicAdd(pd + 1, wp * rel.y);
                        atomicAdd(pd + 2, wp * rel.z);
                    }
                }
        }
    }
}

// ---------------------------------------------------------------------------
extern "C" void kernel_launch(void* const* d_in, const int* in_sizes, int n_in,
                              void* d_out, int out_size, void* d_ws, size_t ws_size,
                              hipStream_t stream) {
    const float* x    = (const float*)d_in[0];
    const float* pos  = (const float*)d_in[1];
    const int*   eidx = (const int*)  d_in[2];
    const float* W1x  = (const float*)d_in[3];
    const float* b1x  = (const float*)d_in[4];
    const float* W2x  = (const float*)d_in[5];
    const float* b2x  = (const float*)d_in[6];
    const float* W1p  = (const float*)d_in[7];
    const float* b1p  = (const float*)d_in[8];
    const float* W2p  = (const float*)d_in[9];
    const float* b2p  = (const float*)d_in[10];

    const int N = in_sizes[0] / F_DIM;
    const int E = in_sizes[2] / 2;

    // workspace layout (all 16B-aligned)
    char* ws = (char*)d_ws;
    unsigned short* W1bf   = (unsigned short*)(ws + 0);        // 128*256*2 = 65536
    unsigned short* W2bf   = (unsigned short*)(ws + 65536);    // 256*80*2  = 40960
    float*          w1last = (float*)(ws + 106496);            // 256*4
    float*          b1cat  = (float*)(ws + 107520);            // 256*4
    float*          b2cat  = (float*)(ws + 108544);            // 80*4

    float* outx = (float*)d_out;
    float* outp = outx + (size_t)N * F_DIM;

    const int ztot = N * (F_DIM + 3);
    zero_f32<<<(ztot + 255) / 256, 256, 0, stream>>>(outx, ztot);
    pack_w1<<<(129 * 256 + 255) / 256, 256, 0, stream>>>(W1x, W1p, b1x, b1p,
                                                         W1bf, w1last, b1cat);
    pack_w2<<<(256 * 80 + 255) / 256, 256, 0, stream>>>(W2x, W2p, b2x, b2p,
                                                        W2bf, b2cat);

    const int edgesPerBlock = TTILE * 16;                      // 64
    const int blocks = (E + edgesPerBlock - 1) / edgesPerBlock;
    egnn_kernel<<<blocks, 32, 0, stream>>>(
        x, pos, eidx, E, N, W1bf, w1last, b1cat, W2bf, b2cat, outx, outp);
}
